// EmbeddingBlock_87411174408854
// MI455X (gfx1250) — matrix operand order repaired and checked by reading, not compiled, not run on hardware
//
#include <hip/hip_runtime.h>

#define H     128
#define NR    6
#define NEMB  95

typedef __attribute__((ext_vector_type(16))) _Float16 v16h;
typedef __attribute__((ext_vector_type(8)))  float    v8f;
typedef __attribute__((ext_vector_type(2)))  float    f2v;

union H16 { v16h v; uint4 q[2]; };

// silu via fast hardware reciprocal: v_exp_f32 + v_add + v_rcp_f32 + v_mul
__device__ __forceinline__ float silu_f(float x) {
    return x * __builtin_amdgcn_rcpf(1.0f + __expf(-x));
}

// workspace layout (bytes) — total 138240 B
#define WS_T1    0          // 95*128 f32
#define WS_T2    48640      // 95*128 f32
#define WS_W3    97280      // 16384 f16  (packed B tiles for W_lin[256:384])
#define WS_WR    130048     // 4096  f16  (packed B tiles for W_rbf, K padded to 32)
#define WS_TOTAL 138240

// ---------- prep 1: T1 = emb @ W_lin[0:128], T2 = emb @ W_lin[128:256] ----------
__global__ void prep_tables(const float* __restrict__ emb, const float* __restrict__ Wlin,
                            float* __restrict__ T1, float* __restrict__ T2) {
    int r = blockIdx.x;      // 0..94
    int n = threadIdx.x;     // 0..127
    float a1 = 0.f, a2 = 0.f;
    for (int k = 0; k < H; ++k) {
        float e = emb[r * H + k];
        a1 += e * Wlin[k * H + n];
        a2 += e * Wlin[(H + k) * H + n];
    }
    T1[r * H + n] = a1;
    T2[r * H + n] = a2;
}

// ---------- prep 2: pack W3 / W_rbf into WMMA 16-bit B-matrix lane layout ----------
// B tile (32K x 16N): lane L -> n = nt*16 + (L&15), kg = L>>4; half h (0..15): K = kg*16 + h
__global__ void prep_pack(const float* __restrict__ Wlin, const float* __restrict__ Wrbf,
                          _Float16* __restrict__ W3p, _Float16* __restrict__ WRp) {
    int idx = blockIdx.x * blockDim.x + threadIdx.x;
    if (idx < 16384) {
        int h = idx & 15, c = idx >> 4;
        int lane = c & 31, nt = (c >> 5) & 7, kt = c >> 8;
        int kg = lane >> 4;
        int n = nt * 16 + (lane & 15);
        int K = kt * 32 + kg * 16 + h;
        W3p[idx] = (_Float16)Wlin[(256 + K) * H + n];
    } else if (idx < 20480) {
        int id2 = idx - 16384;
        int h = id2 & 15, c = id2 >> 4;
        int lane = c & 31, nt = c >> 5;
        int kg = lane >> 4;
        int n = nt * 16 + (lane & 15);
        int K = kg * 16 + h;
        WRp[id2] = (K < NR) ? (_Float16)Wrbf[K * H + n] : (_Float16)0.0f;
    }
}

// ---------- LDS load helpers ----------
__device__ __forceinline__ v16h ld32B(const _Float16* p) {
    H16 u;
    u.q[0] = *(const uint4*)p;
    u.q[1] = *(const uint4*)(p + 8);
    return u.v;
}
__device__ __forceinline__ v16h ld2x16B(const _Float16* p0, const _Float16* p1) {
    H16 u;
    u.q[0] = *(const uint4*)p0;
    u.q[1] = *(const uint4*)p1;
    return u.v;
}

// ---------- main fused kernel: 8 waves/block, 16 edges per wave per tile ----------
// LDS (dynamic, 173056 B):
//  [0)       sT1   48640
//  [48640)   sT2   48640
//  [97280)   sW3   32768
//  [130048)  sWR    8192
//  [138240)  sBL     512
//  [138752)  sBR     512
//  [139264)  sIdx   1024   (8 waves x 16 edges x 2 ints)
//  [140288)  sStg  32768   (8 waves x 16 x 128 f16 rbf_h staging)
__global__ __launch_bounds__(256)
void fused_edge_kernel(const int* __restrict__ xA, const float* __restrict__ rbf,
                       const int* __restrict__ iA, const int* __restrict__ jA,
                       const float* __restrict__ bR, const float* __restrict__ bL,
                       const char* __restrict__ ws, float* __restrict__ out, int nE) {
    extern __shared__ char smem[];
    float*    sT1 = (float*)(smem);
    float*    sT2 = (float*)(smem + 48640);
    _Float16* sW3 = (_Float16*)(smem + 97280);
    _Float16* sWR = (_Float16*)(smem + 130048);
    float*    sBL = (float*)(smem + 138240);
    float*    sBR = (float*)(smem + 138752);
    int*      sIdx = (int*)(smem + 139264);
    _Float16* sStg = (_Float16*)(smem + 140288);

    // one-time fill: T1,T2,W3p,WRp are contiguous in ws with identical layout
    {
        const uint4* src = (const uint4*)ws;
        uint4* dst = (uint4*)smem;
        for (int t = threadIdx.x; t < (WS_TOTAL / 16); t += blockDim.x) dst[t] = src[t];
        if (threadIdx.x < 128)       sBL[threadIdx.x]       = bL[threadIdx.x];
        else if (threadIdx.x < 256)  sBR[threadIdx.x - 128] = bR[threadIdx.x - 128];
        __syncthreads();
    }

    const int lane = threadIdx.x & 31;
    const int wave = threadIdx.x >> 5;
    const int m  = lane & 15;    // edge row within tile (A-layout row)
    const int kg = lane >> 4;    // lane K-group
    const int n0 = lane & 15;    // output column within N-tile (B/C layout)

    int*      myIdx = sIdx + wave * 32;     // [16][2]
    _Float16* stg   = sStg + wave * 2048;   // [16][128] f16

    const int ntiles = (nE + 15) >> 4;
    const int nWaves = gridDim.x * 8;

    for (int tile = blockIdx.x * 8 + wave; tile < ntiles; tile += nWaves) {
        int e  = tile * 16 + m;
        int ec = (e < nE) ? e : (nE - 1);

        // edge -> embedding-row indices, broadcast through LDS
        if (kg == 0) {
            int ei = __builtin_nontemporal_load(iA + ec);
            int ej = __builtin_nontemporal_load(jA + ec);
            myIdx[m * 2 + 0] = xA[ei];
            myIdx[m * 2 + 1] = xA[ej];
        }

        // ---- build A for rbf GEMM (16 edges x 32K, K>=6 zero; lanes kg=1 all zero) ----
        H16 ua;
        ua.q[0] = make_uint4(0u, 0u, 0u, 0u);
        ua.q[1] = make_uint4(0u, 0u, 0u, 0u);
        if (kg == 0) {
            const f2v* rp = (const f2v*)(rbf + (size_t)ec * NR);
            f2v f01 = __builtin_nontemporal_load(rp + 0);
            f2v f23 = __builtin_nontemporal_load(rp + 1);
            f2v f45 = __builtin_nontemporal_load(rp + 2);
            ua.v[0] = (_Float16)f01.x; ua.v[1] = (_Float16)f01.y;
            ua.v[2] = (_Float16)f23.x; ua.v[3] = (_Float16)f23.y;
            ua.v[4] = (_Float16)f45.x; ua.v[5] = (_Float16)f45.y;
        }
        v16h aR = ua.v;

        // ---- phase 1: rbf_h = silu(rbf @ W_rbf + b_rbf) -> per-wave LDS staging ----
#pragma unroll
        for (int nt = 0; nt < 8; ++nt) {
            v16h b = ld32B(sWR + (nt * 32 + lane) * 16);
            float br = sBR[nt * 16 + n0];
            v8f c = { br, br, br, br, br, br, br, br };
            c = __builtin_amdgcn_wmma_f32_16x16x32_f16(false, aR, false, b,
                                                       (short)0, c, false, false);
#pragma unroll
            for (int r = 0; r < 8; ++r)
                stg[(kg * 8 + r) * H + nt * 16 + n0] = (_Float16)silu_f(c[r]);
        }

        // ---- reload rbf_h in A-layout (same wave -> DS-count dependency only) ----
        v16h aS[4];
#pragma unroll
        for (int kt = 0; kt < 4; ++kt)
            aS[kt] = ld2x16B(stg + m * H + kt * 32 + kg * 16,
                             stg + m * H + kt * 32 + kg * 16 + 8);

        int ri[8], rj[8];
#pragma unroll
        for (int r = 0; r < 8; ++r) {
            ri[r] = myIdx[(kg * 8 + r) * 2 + 0];
            rj[r] = myIdx[(kg * 8 + r) * 2 + 1];
        }

        // ---- phase 2: out = silu(T1[xi] + T2[xj] + rbf_h @ W3 + b_lin) ----
#pragma unroll 2
        for (int nt = 0; nt < 8; ++nt) {
            int n = nt * 16 + n0;
            float bl = sBL[n];
            v8f c;
#pragma unroll
            for (int r = 0; r < 8; ++r)
                c[r] = bl + sT1[ri[r] * H + n] + sT2[rj[r] * H + n];
#pragma unroll
            for (int kt = 0; kt < 4; ++kt) {
                v16h b = ld32B(sW3 + ((kt * 8 + nt) * 32 + lane) * 16);
                c = __builtin_amdgcn_wmma_f32_16x16x32_f16(false, aS[kt], false, b,
                                                           (short)0, c, false, false);
            }
#pragma unroll
            for (int r = 0; r < 8; ++r) {
                int er = tile * 16 + kg * 8 + r;
                if (er < nE)
                    __builtin_nontemporal_store(silu_f(c[r]), out + (size_t)er * H + n);
            }
        }
    }
}

extern "C" void kernel_launch(void* const* d_in, const int* in_sizes, int n_in,
                              void* d_out, int out_size, void* d_ws, size_t ws_size,
                              hipStream_t stream) {
    const int*   x    = (const int*)  d_in[0];
    const float* rbf  = (const float*)d_in[1];
    const int*   iA   = (const int*)  d_in[2];
    const int*   jA   = (const int*)  d_in[3];
    const float* emb  = (const float*)d_in[4];
    const float* Wrbf = (const float*)d_in[5];
    const float* bR   = (const float*)d_in[6];
    const float* Wlin = (const float*)d_in[7];
    const float* bL   = (const float*)d_in[8];
    float* out = (float*)d_out;
    const int nE = in_sizes[2];

    char* ws = (char*)d_ws;  // needs WS_TOTAL = 138240 bytes
    float*    T1  = (float*)(ws + WS_T1);
    float*    T2  = (float*)(ws + WS_T2);
    _Float16* W3p = (_Float16*)(ws + WS_W3);
    _Float16* WRp = (_Float16*)(ws + WS_WR);

    prep_tables<<<NEMB, H, 0, stream>>>(emb, Wlin, T1, T2);
    prep_pack<<<80, 256, 0, stream>>>(Wlin, Wrbf, W3p, WRp);

    const int nBlocks = 1280;
    const size_t shmem = 173056;
    fused_edge_kernel<<<nBlocks, 256, shmem, stream>>>(x, rbf, iA, jA, bR, bL,
                                                       (const char*)ws, out, nE);
}
